// MultiHeadAttention_14800457302227
// MI455X (gfx1250) — matrix operand order repaired and checked by reading, not compiled, not run on hardware
//
#include <hip/hip_runtime.h>
#include <hip/hip_bf16.h>

typedef _Float16 v16h __attribute__((ext_vector_type(16)));
typedef _Float16 v8h  __attribute__((ext_vector_type(8)));
typedef float    v8f  __attribute__((ext_vector_type(8)));
typedef unsigned int u32x4 __attribute__((ext_vector_type(4)));
typedef int i32x4 __attribute__((ext_vector_type(4)));
typedef int i32x8 __attribute__((ext_vector_type(8)));

#define B_  4
#define T_  2048
#define C_  1024
#define H_  16
#define D_  64
#define M_  (B_ * T_)

#if __has_builtin(__builtin_amdgcn_tensor_load_to_lds) && __has_builtin(__builtin_amdgcn_s_wait_tensorcnt)
#define USE_TDM 1
#else
#define USE_TDM 0
#endif

// A-operand (16x32 f16) per-lane load, contiguous K order:
// lane half 0: K = k0+0..15; half 1: K = k0+16..31.
__device__ __forceinline__ v16h load_a16x32(const _Float16* rowbase, int ld,
                                            int k0, int ln, int half) {
  const _Float16* p = rowbase + (size_t)ln * ld + k0 + half * 16;
  v8h lo = *(const v8h*)(p);
  v8h hi = *(const v8h*)(p + 8);
  v16h r;
#pragma unroll
  for (int i = 0; i < 8; ++i) { r[i] = lo[i]; r[i + 8] = hi[i]; }
  return r;
}

__device__ __forceinline__ v8f wmma_f16(v16h a, v16h b, v8f c) {
  return __builtin_amdgcn_wmma_f32_16x16x32_f16(false, a, false, b, (short)0, c,
                                                false, false);
}

#if USE_TDM
// Issue a TDM 2D tile load (tile_rows x tile_cols of f16) into LDS.
// Descriptor packing per CDNA5 ISA 8.3/8.4 (2D tensor: groups 2/3 zero).
__device__ __forceinline__ void tdm_load_tile_f16(const _Float16* gbase,
                                                  unsigned lds_byte_off,
                                                  int tile_rows, int tile_cols,
                                                  int row_stride_elems) {
  unsigned long long ga = (unsigned long long)(size_t)gbase;
  u32x4 g0;
  g0[0] = 1u;                                           // count=1, user desc
  g0[1] = lds_byte_off;                                 // lds_addr (bytes)
  g0[2] = (unsigned)ga;                                 // global_addr[31:0]
  g0[3] = (unsigned)((ga >> 32) & 0x1FFFFFFull) | (2u << 30);  // addr hi | type=2
  i32x8 g1;
  g1[0] = 1 << 16;                                      // data_size=1 (2B), mask=0
  g1[1] = (tile_cols & 0xFFFF) << 16;                   // tensor_dim0 lo @ [63:48]
  g1[2] = (tile_rows & 0xFFFF) << 16;                   // tensor_dim1 lo @ [95:80]
  g1[3] = (tile_cols & 0xFFFF) << 16;                   // tile_dim0 @ [127:112]
  g1[4] = tile_rows & 0xFFFF;                           // tile_dim1 @ [143:128]
  g1[5] = row_stride_elems;                             // tensor_dim0_stride lo
  g1[6] = 0;
  g1[7] = 0;
  i32x4 z4 = {0, 0, 0, 0};
#if __clang_major__ >= 23
  i32x8 z8 = {0, 0, 0, 0, 0, 0, 0, 0};
  __builtin_amdgcn_tensor_load_to_lds(g0, g1, z4, z4, z8, 0);
#else
  __builtin_amdgcn_tensor_load_to_lds(g0, g1, z4, z4, 0);
#endif
}
#else
// Fallback: cooperative 128x32 f16 tile load, 256 threads x 32B.
__device__ __forceinline__ void coop_load_tile_f16(const _Float16* gbase,
                                                   _Float16* dst,
                                                   int row_stride_elems, int tid) {
  int row = tid >> 1;
  int seg = (tid & 1) * 16;
  *(v16h*)(dst + row * 32 + seg) =
      *(const v16h*)(gbase + (size_t)row * row_stride_elems + seg);
}
#endif

__global__ void f32_to_f16_kernel(const float* __restrict__ src,
                                  _Float16* __restrict__ dst, int n) {
  int i = blockIdx.x * blockDim.x + threadIdx.x;
  int stride = gridDim.x * blockDim.x;
  for (; i < n; i += stride) dst[i] = (_Float16)src[i];
}

// y = A[M_,C_] @ W[C_,C_]^T (+bias) * out_scale, 128x128 block tile,
// double-buffered TDM-staged LDS panels, 8 waves x (32x64) per block.
// mode 0: f16 out [B,H,T,D]; mode 1: f16 out [B,H,D,T]; mode 2: f32 [M_,C_]
__global__ void __launch_bounds__(256)
gemm_wmma_kernel(const _Float16* __restrict__ A, const _Float16* __restrict__ W,
                 const float* __restrict__ bias, void* __restrict__ out,
                 float out_scale, int mode) {
  const int tid  = threadIdx.x;
  const int lane = tid & 31;
  const int warp = tid >> 5;
  const int ln = lane & 15, half = lane >> 4;
  const int nblkcnt = C_ / 128;                 // 8
  const int m0 = (blockIdx.x / nblkcnt) * 128;
  const int n0 = (blockIdx.x % nblkcnt) * 128;
  const int wm = warp >> 1;                     // 0..3: 32-row strip
  const int wn = warp & 1;                      // 0..1: 64-col strip

  __shared__ _Float16 Abuf[2][128 * 32];        // 8 KB each
  __shared__ _Float16 Bbuf[2][128 * 32];

  const _Float16* Ag = A + (size_t)m0 * C_;
  const _Float16* Wg = W + (size_t)n0 * C_;

#if USE_TDM
  if (warp == 0) {
    tdm_load_tile_f16(Ag + 0,  (unsigned)(size_t)&Abuf[0][0], 128, 32, C_);
    tdm_load_tile_f16(Wg + 0,  (unsigned)(size_t)&Bbuf[0][0], 128, 32, C_);
    tdm_load_tile_f16(Ag + 32, (unsigned)(size_t)&Abuf[1][0], 128, 32, C_);
    tdm_load_tile_f16(Wg + 32, (unsigned)(size_t)&Bbuf[1][0], 128, 32, C_);
  }
#else
  coop_load_tile_f16(Ag + 0,  Abuf[0], C_, tid);
  coop_load_tile_f16(Wg + 0,  Bbuf[0], C_, tid);
  coop_load_tile_f16(Ag + 32, Abuf[1], C_, tid);
  coop_load_tile_f16(Wg + 32, Bbuf[1], C_, tid);
#endif

  v8f acc[2][4] = {};
  const int KSTEPS = C_ / 32;                   // 32

  for (int i = 0; i < KSTEPS; ++i) {
#if USE_TDM
    if (warp == 0) {
      if (i < KSTEPS - 1) __builtin_amdgcn_s_wait_tensorcnt(2);
      else                __builtin_amdgcn_s_wait_tensorcnt(0);
    }
#endif
    __syncthreads();

    const _Float16* Ab = Abuf[i & 1] + (wm * 32) * 32;
    const _Float16* Bb = Bbuf[i & 1] + (wn * 64) * 32;
    // issue all 6 LDS loads up front (distinct registers), then drain WMMAs
    v16h a0 = load_a16x32(Ab, 32, 0, ln, half);
    v16h a1 = load_a16x32(Ab + 16 * 32, 32, 0, ln, half);
    v16h bm[4];
#pragma unroll
    for (int j = 0; j < 4; ++j)
      bm[j] = *(const v16h*)(Bb + (size_t)(j * 16 + ln) * 32 + 16 * half);
#pragma unroll
    for (int j = 0; j < 4; ++j) {
      acc[0][j] = wmma_f16(a0, bm[j], acc[0][j]);
      acc[1][j] = wmma_f16(a1, bm[j], acc[1][j]);
    }

    __syncthreads();
    if (i + 2 < KSTEPS) {
      const int knext = (i + 2) * 32;
#if USE_TDM
      if (warp == 0) {
        tdm_load_tile_f16(Ag + knext, (unsigned)(size_t)&Abuf[i & 1][0], 128, 32, C_);
        tdm_load_tile_f16(Wg + knext, (unsigned)(size_t)&Bbuf[i & 1][0], 128, 32, C_);
      }
#else
      coop_load_tile_f16(Ag + knext, Abuf[i & 1], C_, tid);
      coop_load_tile_f16(Wg + knext, Bbuf[i & 1], C_, tid);
#endif
    }
  }

#pragma unroll
  for (int sm = 0; sm < 2; ++sm) {
#pragma unroll
    for (int j = 0; j < 4; ++j) {
      const int n = n0 + wn * 64 + j * 16 + ln;
      const float bj = bias ? bias[n] : 0.0f;
#pragma unroll
      for (int r = 0; r < 8; ++r) {
        const int m = m0 + wm * 32 + sm * 16 + r + 8 * half;  // C/D layout row
        const float v = (acc[sm][j][r] + bj) * out_scale;
        const int bb = m / T_, t = m % T_;
        const int hh = n / D_, d = n % D_;
        if (mode == 0) {
          ((_Float16*)out)[(((size_t)bb * H_ + hh) * T_ + t) * D_ + d] = (_Float16)v;
        } else if (mode == 1) {
          ((_Float16*)out)[(((size_t)bb * H_ + hh) * D_ + d) * T_ + t] = (_Float16)v;
        } else {
          ((float*)out)[(size_t)m * C_ + n] = v;
        }
      }
    }
  }
}

// Flash attention: one wave per 16-row Q tile; K blocks of 32 keys.
// Q,K: [B,H,T,D] f16 (Q pre-scaled by 1/sqrt(D)); Vt: [B,H,D,T] f16.
__global__ void __launch_bounds__(128)
flash_attn_kernel(const _Float16* __restrict__ Qh, const _Float16* __restrict__ Kh,
                  const _Float16* __restrict__ Vth, _Float16* __restrict__ attn) {
  const int lane = threadIdx.x & 31;
  const int warp = threadIdx.x >> 5;
  const int wid  = blockIdx.x * 4 + warp;
  const int qtiles = T_ / 16;
  const int bh = wid / qtiles;
  const int q0 = (wid % qtiles) * 16;
  const int b  = bh / H_;
  const int h  = bh % H_;
  const int ln = lane & 15, half = lane >> 4;

  const _Float16* Qp = Qh  + ((size_t)bh * T_ + q0) * D_;
  const _Float16* Kp = Kh  + (size_t)bh * T_ * D_;
  const _Float16* Vp = Vth + (size_t)bh * D_ * T_;

  __shared__ _Float16 plds_all[4][16 * 32];
  _Float16* plds = plds_all[warp];

  const v16h aq0 = load_a16x32(Qp, D_, 0,  ln, half);
  const v16h aq1 = load_a16x32(Qp, D_, 32, ln, half);

  float rm[8], rl[8];
#pragma unroll
  for (int r = 0; r < 8; ++r) { rm[r] = -1e30f; rl[r] = 0.0f; }
  v8f o[4] = {};

  for (int kb = 0; kb < T_; kb += 32) {
    // prefetch next key/value block (global_prefetch_b8)
    if (kb + 32 < T_) {
      __builtin_prefetch((const void*)(Kp + (size_t)(kb + 32 + lane) * D_), 0, 1);
      __builtin_prefetch((const void*)(Vp + (size_t)(lane * 2) * T_ + kb + 32), 0, 1);
    }
    // scores: two 16x16 tiles (keys kb..kb+15, kb+16..kb+31)
    v8f s[2];
#pragma unroll
    for (int j = 0; j < 2; ++j) {
      const _Float16* kcol = Kp + (size_t)(kb + j * 16 + ln) * D_ + 16 * half;
      v16h b0 = *(const v16h*)(kcol);
      v16h b1 = *(const v16h*)(kcol + 32);
      v8f sc = {};
      sc = wmma_f16(aq0, b0, sc);
      sc = wmma_f16(aq1, b1, sc);
      s[j] = sc;
    }
    // online softmax over this 32-key block (row = r + 8*half, col = lane&15)
#pragma unroll
    for (int r = 0; r < 8; ++r) {
      float x0 = s[0][r], x1 = s[1][r];
      float mx = fmaxf(x0, x1);
#pragma unroll
      for (int off = 1; off < 16; off <<= 1) mx = fmaxf(mx, __shfl_xor(mx, off, 16));
      float mnew = fmaxf(rm[r], mx);
      float corr = __expf(rm[r] - mnew);
      float p0 = __expf(x0 - mnew);
      float p1 = __expf(x1 - mnew);
      float ps = p0 + p1;
#pragma unroll
      for (int off = 1; off < 16; off <<= 1) ps += __shfl_xor(ps, off, 16);
      rl[r] = rl[r] * corr + ps;
      rm[r] = mnew;
#pragma unroll
      for (int j2 = 0; j2 < 4; ++j2) o[j2][r] *= corr;
      plds[(r + 8 * half) * 32 + ln]      = (_Float16)p0;
      plds[(r + 8 * half) * 32 + 16 + ln] = (_Float16)p1;
    }
    asm volatile("s_wait_dscnt 0" ::: "memory");
    // reload P as a 16x32 A operand
    v16h pA;
    {
      const _Float16* pr = plds + ln * 32 + half * 16;
      v8h lo = *(const v8h*)(pr);
      v8h hi = *(const v8h*)(pr + 8);
#pragma unroll
      for (int i = 0; i < 8; ++i) { pA[i] = lo[i]; pA[i + 8] = hi[i]; }
    }
    // O += P @ V
#pragma unroll
    for (int j2 = 0; j2 < 4; ++j2) {
      const _Float16* vcol = Vp + (size_t)(j2 * 16 + ln) * T_ + kb + 16 * half;
      v16h bv = *(const v16h*)(vcol);
      o[j2] = wmma_f16(pA, bv, o[j2]);
    }
  }

#pragma unroll
  for (int r = 0; r < 8; ++r) {
    float inv = 1.0f / rl[r];
    int t = q0 + r + 8 * half;
    size_t rowbase = ((size_t)b * T_ + t) * C_ + (size_t)h * D_;
#pragma unroll
    for (int j2 = 0; j2 < 4; ++j2)
      attn[rowbase + j2 * 16 + ln] = (_Float16)(o[j2][r] * inv);
  }
}

extern "C" void kernel_launch(void* const* d_in, const int* in_sizes, int n_in,
                              void* d_out, int out_size, void* d_ws, size_t ws_size,
                              hipStream_t stream) {
  const float* x  = (const float*)d_in[0];
  const float* Wq = (const float*)d_in[1];
  const float* bq = (const float*)d_in[2];
  const float* Wk = (const float*)d_in[3];
  const float* Wv = (const float*)d_in[4];
  const float* bv = (const float*)d_in[5];
  const float* Wo = (const float*)d_in[6];
  const float* bo = (const float*)d_in[7];

  char* ws = (char*)d_ws;
  const size_t MB = 1ull << 20;
  _Float16* xh  = (_Float16*)(ws +  0 * MB);  // 16 MB  x in f16
  _Float16* Wqh = (_Float16*)(ws + 16 * MB);  // 2 MB
  _Float16* Wkh = (_Float16*)(ws + 18 * MB);  // 2 MB
  _Float16* Wvh = (_Float16*)(ws + 20 * MB);  // 2 MB
  _Float16* Woh = (_Float16*)(ws + 22 * MB);  // 2 MB
  _Float16* Qh  = (_Float16*)(ws + 24 * MB);  // 16 MB [B,H,T,D]
  _Float16* Kh  = (_Float16*)(ws + 40 * MB);  // 16 MB [B,H,T,D]
  _Float16* Vth = (_Float16*)(ws + 56 * MB);  // 16 MB [B,H,D,T]
  _Float16* At  = (_Float16*)(ws + 72 * MB);  // 16 MB [B,T,C]

  f32_to_f16_kernel<<<8192, 256, 0, stream>>>(x,  xh,  M_ * C_);
  f32_to_f16_kernel<<<1024, 256, 0, stream>>>(Wq, Wqh, C_ * C_);
  f32_to_f16_kernel<<<1024, 256, 0, stream>>>(Wk, Wkh, C_ * C_);
  f32_to_f16_kernel<<<1024, 256, 0, stream>>>(Wv, Wvh, C_ * C_);
  f32_to_f16_kernel<<<1024, 256, 0, stream>>>(Wo, Woh, C_ * C_);

  const int gemm_blocks = (M_ / 128) * (C_ / 128);    // 512 blocks x 256 thr
  // Q gets bias + 1/sqrt(D) folded in; K no bias; V transposed layout.
  gemm_wmma_kernel<<<gemm_blocks, 256, 0, stream>>>(xh, Wqh, bq,      Qh,  0.125f, 0);
  gemm_wmma_kernel<<<gemm_blocks, 256, 0, stream>>>(xh, Wkh, nullptr, Kh,  1.0f,   0);
  gemm_wmma_kernel<<<gemm_blocks, 256, 0, stream>>>(xh, Wvh, bv,      Vth, 1.0f,   1);

  const int fa_blocks = (B_ * H_ * (T_ / 16)) / 4;    // 2048 blocks x 4 waves
  flash_attn_kernel<<<fa_blocks, 128, 0, stream>>>(Qh, Kh, Vth, At);

  gemm_wmma_kernel<<<gemm_blocks, 256, 0, stream>>>(At, Woh, bo, d_out, 1.0f, 2);
}
